// Attention_1769526526029
// MI455X (gfx1250) — matrix operand order repaired and checked by reading, not compiled, not run on hardware
//
#include <hip/hip_runtime.h>
#include <hip/hip_bf16.h>

typedef __attribute__((ext_vector_type(16))) _Float16 v16h;
typedef __attribute__((ext_vector_type(8)))  float    v8f;

#define WSZ     8
#define NH      4
#define DIM     128
#define NTOK    64
#define HD      32
#define THREADS 256

// wswz layout (f16, in d_ws):
//   [0, 49152)      : w_qkv fragments  ((nt*4+ksb)*32 + lane)*16 halves, nt=0..23, ksb=0..3
//   [49152, 65536)  : w_proj fragments ((nt*4+ksb)*32 + lane)*16 halves, nt=0..7,  ksb=0..3
#define WQ_HALVES 49152
#define WP_HALVES 16384

union AFrag {
    v16h v;
    unsigned int u[8];
    uint4 q[2];
    _Float16 h[16];
};

// ---------- pre-pass: convert + swizzle weights into WMMA B-fragment order ----------
__global__ __launch_bounds__(256) void pack_weights(
    const float* __restrict__ w_qkv,     // [128,384]
    const float* __restrict__ w_proj,    // [128,128]
    _Float16* __restrict__ wswz)
{
    int p = blockIdx.x * 256 + threadIdx.x;        // 0..32767 (pairs of halves)
    union { _Float16 h[2]; unsigned int u; } pk;
    if (p < 24576) {                               // w_qkv region
        int j = p & 7, lane = (p >> 3) & 31, ksb = (p >> 8) & 3, nt = p >> 10;
        int kb   = ksb * 32 + 2 * j + ((lane >= 16) ? 16 : 0);
        int ncol = nt * 16 + (lane & 15);
        pk.h[0] = (_Float16)w_qkv[kb * 384 + ncol];
        pk.h[1] = (_Float16)w_qkv[(kb + 1) * 384 + ncol];
        *(unsigned int*)&wswz[p * 2] = pk.u;
    } else {                                       // w_proj region
        int qi = p - 24576;                        // 0..8191
        int j = qi & 7, lane = (qi >> 3) & 31, ksb = (qi >> 8) & 3, nt = qi >> 10;
        int kb   = ksb * 32 + 2 * j + ((lane >= 16) ? 16 : 0);
        int ncol = nt * 16 + (lane & 15);
        pk.h[0] = (_Float16)w_proj[kb * 128 + ncol];
        pk.h[1] = (_Float16)w_proj[(kb + 1) * 128 + ncol];
        *(unsigned int*)&wswz[WQ_HALVES + qi * 2] = pk.u;
    }
}

__global__ __launch_bounds__(THREADS) void swin_window_attn(
    const float* __restrict__ x,         // [4096,64,128]
    const _Float16* __restrict__ wswz,   // pre-swizzled f16 weights (d_ws)
    const float* __restrict__ b_proj,    // [128]
    const float* __restrict__ bias_tab,  // [225,4]
    float* __restrict__ out)             // [4096,64,128]
{
    // 64KB LDS, phase-aliased:
    //  [0,16K)   : xs[64][128] f16 (A)  -> bias_s (B) -> ys[64][128] f16 (C->D)
    //  [16K,48K) : qs[64][128], ks[64][128] f16 (A->B)  -> ps[4][64][64] f16 (B->C)
    //  [48K,64K) : vt[128][64] f16 (A->C), V stored transposed
    __shared__ char smem[65536];
    _Float16* regA = (_Float16*)smem;
    _Float16* qs   = (_Float16*)(smem + 16384);
    _Float16* ks   = (_Float16*)(smem + 32768);
    _Float16* ps   = (_Float16*)(smem + 16384);
    _Float16* vt   = (_Float16*)(smem + 49152);
    float*    bias_s = (float*)smem;

    const int b     = blockIdx.x;
    const int tid   = threadIdx.x;
    const int w     = tid >> 5;
    const int lane  = tid & 31;
    const int lhalf = (lane >= 16) ? 1 : 0;
    const int l15   = lane & 15;

    const uint4* wq4 = (const uint4*)wswz;                  // b128 fragment units
    const uint4* wp4 = (const uint4*)(wswz + WQ_HALVES);

    // ---------------- stage x -> LDS (f16), vectorized b128 loads ----------------
    const float4* xb4 = (const float4*)(x + (size_t)b * (NTOK * DIM));
    #pragma unroll
    for (int i = 0; i < 8; ++i) {
        int idx = i * THREADS + tid;            // 0..2047 float4s
        float4 v = xb4[idx];
        union { _Float16 h[4]; unsigned long long u; } pk;
        pk.h[0] = (_Float16)v.x; pk.h[1] = (_Float16)v.y;
        pk.h[2] = (_Float16)v.z; pk.h[3] = (_Float16)v.w;
        *(unsigned long long*)&regA[idx * 4] = pk.u;
    }
    __syncthreads();

    // ---------------- Phase A: QKV = x @ w_qkv  ([64,128]x[128,384]) ----------------
    // wave w owns N-tiles {3w,3w+1,3w+2}; B-fragments are 2 coalesced b128 loads each
    for (int nti = 0; nti < 3; ++nti) {
        const int nt   = w * 3 + nti;
        const int ncol = nt * 16 + l15;         // 0..383
        AFrag bfr[4];
        #pragma unroll
        for (int ksb = 0; ksb < 4; ++ksb) {
            int base = ((nt * 4 + ksb) * 32 + lane) * 2;    // uint4 units
            bfr[ksb].q[0] = wq4[base];
            bfr[ksb].q[1] = wq4[base + 1];
        }
        for (int mt = 0; mt < 4; ++mt) {
            v8f acc = {};
            const int m = mt * 16 + l15;
            #pragma unroll
            for (int ksb = 0; ksb < 4; ++ksb) {
                AFrag a;
                #pragma unroll
                for (int j = 0; j < 8; ++j) {
                    int klo = ksb * 32 + 2 * j + (lhalf ? 16 : 0);
                    a.u[j] = *(const unsigned int*)&regA[m * DIM + klo];
                }
                acc = __builtin_amdgcn_wmma_f32_16x16x32_f16(
                    false, a.v, false, bfr[ksb].v, (short)0, acc, false, false);
            }
            if (nt < 8) {                        // Q tile
                #pragma unroll
                for (int r = 0; r < 8; ++r) {
                    int mrow = mt * 16 + r + (lhalf ? 8 : 0);
                    qs[mrow * DIM + ncol] = (_Float16)acc[r];
                }
            } else if (nt < 16) {                // K tile
                #pragma unroll
                for (int r = 0; r < 8; ++r) {
                    int mrow = mt * 16 + r + (lhalf ? 8 : 0);
                    ks[mrow * DIM + (ncol - 128)] = (_Float16)acc[r];
                }
            } else {                             // V tile -> transposed, packed b128 store
                union { _Float16 h[8]; uint4 q; } st;
                #pragma unroll
                for (int r = 0; r < 8; ++r) st.h[r] = (_Float16)acc[r];
                *(uint4*)&vt[(ncol - 256) * NTOK + mt * 16 + lhalf * 8] = st.q;
            }
        }
    }
    __syncthreads();

    // stage rel-pos bias table (900 floats) into the now-dead x region
    for (int i = tid; i < 225 * NH; i += THREADS) bias_s[i] = bias_tab[i];
    __syncthreads();

    // ---------------- Phase B: scores + softmax (2 waves per head) ----------------
    const int h     = w >> 1;
    const int mhalf = w & 1;
    const float scale = 0.17677669529663687f;   // 1/sqrt(32)
    v8f prob[2][4];

    // hoist K^T B-fragments (reused by both local mtiles)
    AFrag kfr[4];
    #pragma unroll
    for (int nt = 0; nt < 4; ++nt) {
        const int nk = nt * 16 + l15;
        #pragma unroll
        for (int j = 0; j < 8; ++j) {
            int kb = 2 * j + (lhalf ? 16 : 0);
            kfr[nt].u[j] = *(const unsigned int*)&ks[nk * DIM + h * HD + kb];
        }
    }

    for (int lmt = 0; lmt < 2; ++lmt) {
        const int mt = mhalf * 2 + lmt;
        const int m  = mt * 16 + l15;
        AFrag a;                                 // Q fragment, K(=hd)=32 -> one WMMA
        #pragma unroll
        for (int j = 0; j < 8; ++j) {
            int klo = 2 * j + (lhalf ? 16 : 0);
            a.u[j] = *(const unsigned int*)&qs[m * DIM + h * HD + klo];
        }
        #pragma unroll
        for (int nt = 0; nt < 4; ++nt) {
            v8f acc = {};
            acc = __builtin_amdgcn_wmma_f32_16x16x32_f16(
                false, a.v, false, kfr[nt].v, (short)0, acc, false, false);
            #pragma unroll
            for (int r = 0; r < 8; ++r) {
                int i_row = mt * 16 + r + (lhalf ? 8 : 0);
                int j_col = nt * 16 + l15;
                int bidx  = ((i_row >> 3) - (j_col >> 3) + 7) * 15
                          + ((i_row & 7) - (j_col & 7) + 7);
                acc[r] = acc[r] * scale + bias_s[bidx * NH + h];
            }
            prob[lmt][nt] = acc;
        }
        // row softmax: a row lives in one 16-lane half (fixed r), cols across lanes+ntiles
        #pragma unroll
        for (int r = 0; r < 8; ++r) {
            float rmax = prob[lmt][0][r];
            rmax = fmaxf(rmax, prob[lmt][1][r]);
            rmax = fmaxf(rmax, prob[lmt][2][r]);
            rmax = fmaxf(rmax, prob[lmt][3][r]);
            for (int off = 1; off < 16; off <<= 1)
                rmax = fmaxf(rmax, __shfl_xor(rmax, off, 32));
            float rsum = 0.f;
            #pragma unroll
            for (int nt = 0; nt < 4; ++nt) {
                float e = __expf(prob[lmt][nt][r] - rmax);
                prob[lmt][nt][r] = e;
                rsum += e;
            }
            for (int off = 1; off < 16; off <<= 1)
                rsum += __shfl_xor(rsum, off, 32);
            float rinv = __frcp_rn(rsum);
            #pragma unroll
            for (int nt = 0; nt < 4; ++nt) prob[lmt][nt][r] *= rinv;
        }
    }
    __syncthreads();   // all q/k/bias reads complete; region2/regionA may be overwritten

    // write P (f16) into ps (aliases q/k region)
    #pragma unroll
    for (int lmt = 0; lmt < 2; ++lmt) {
        const int mt = mhalf * 2 + lmt;
        #pragma unroll
        for (int nt = 0; nt < 4; ++nt) {
            #pragma unroll
            for (int r = 0; r < 8; ++r) {
                int i_row = mt * 16 + r + (lhalf ? 8 : 0);
                int j_col = nt * 16 + l15;
                ps[(h * 64 + i_row) * 64 + j_col] = (_Float16)prob[lmt][nt][r];
            }
        }
    }

    // ---------------- Phase C: out_h = P @ V_h ([64,64]x[64,32]) ----------------
    // hoist V B-fragments (reused by both local mtiles)
    AFrag vfr[2][2];                             // [nt2][ksb]
    #pragma unroll
    for (int nt2 = 0; nt2 < 2; ++nt2) {
        const int n = nt2 * 16 + l15;
        #pragma unroll
        for (int ksb = 0; ksb < 2; ++ksb) {
            #pragma unroll
            for (int j = 0; j < 8; ++j) {
                int kb = ksb * 32 + 2 * j + (lhalf ? 16 : 0);
                vfr[nt2][ksb].u[j] =
                    *(const unsigned int*)&vt[(h * HD + n) * NTOK + kb];
            }
        }
    }
    for (int lmt = 0; lmt < 2; ++lmt) {
        const int mt = mhalf * 2 + lmt;
        const int m  = mt * 16 + l15;
        #pragma unroll
        for (int nt2 = 0; nt2 < 2; ++nt2) {
            v8f acc = {};
            #pragma unroll
            for (int ksb = 0; ksb < 2; ++ksb) {
                AFrag a;
                #pragma unroll
                for (int j = 0; j < 8; ++j) {
                    int klo = ksb * 32 + 2 * j + (lhalf ? 16 : 0);
                    a.u[j] = *(const unsigned int*)&ps[(h * 64 + m) * 64 + klo];
                }
                acc = __builtin_amdgcn_wmma_f32_16x16x32_f16(
                    false, a.v, false, vfr[nt2][ksb].v, (short)0, acc, false, false);
            }
            const int n = nt2 * 16 + l15;
            #pragma unroll
            for (int r = 0; r < 8; ++r) {
                int mrow = mt * 16 + r + (lhalf ? 8 : 0);
                regA[mrow * DIM + h * HD + n] = (_Float16)acc[r];   // ys
            }
        }
    }
    __syncthreads();

    // ---------------- Phase D: out = ys @ w_proj + b_proj ----------------
    {
        const int ncol = w * 16 + l15;           // wave w owns N-tile w
        const float bb = b_proj[ncol];
        float* ob = out + (size_t)b * (NTOK * DIM);

        // w_proj B-fragments: 2 coalesced b128 loads each
        AFrag pfr[4];
        #pragma unroll
        for (int ksb = 0; ksb < 4; ++ksb) {
            int base = ((w * 4 + ksb) * 32 + lane) * 2;     // uint4 units
            pfr[ksb].q[0] = wp4[base];
            pfr[ksb].q[1] = wp4[base + 1];
        }
        for (int mt = 0; mt < 4; ++mt) {
            v8f acc = {};
            const int m = mt * 16 + l15;
            #pragma unroll
            for (int ksb = 0; ksb < 4; ++ksb) {
                AFrag a;
                #pragma unroll
                for (int j = 0; j < 8; ++j) {
                    int klo = ksb * 32 + 2 * j + (lhalf ? 16 : 0);
                    a.u[j] = *(const unsigned int*)&regA[m * DIM + klo];
                }
                acc = __builtin_amdgcn_wmma_f32_16x16x32_f16(
                    false, a.v, false, pfr[ksb].v, (short)0, acc, false, false);
            }
            #pragma unroll
            for (int r = 0; r < 8; ++r) {
                int mrow = mt * 16 + r + (lhalf ? 8 : 0);
                ob[mrow * DIM + ncol] = acc[r] + bb;
            }
        }
    }
}

extern "C" void kernel_launch(void* const* d_in, const int* in_sizes, int n_in,
                              void* d_out, int out_size, void* d_ws, size_t ws_size,
                              hipStream_t stream) {
    (void)in_sizes; (void)n_in; (void)out_size; (void)ws_size;
    const float* x        = (const float*)d_in[0];
    const float* w_qkv    = (const float*)d_in[1];
    const float* w_proj   = (const float*)d_in[2];
    const float* b_proj   = (const float*)d_in[3];
    const float* bias_tab = (const float*)d_in[4];
    float* out = (float*)d_out;
    _Float16* wswz = (_Float16*)d_ws;            // needs 131072 bytes

    pack_weights<<<dim3(128), dim3(256), 0, stream>>>(w_qkv, w_proj, wswz);
    swin_window_attn<<<dim3(4096), dim3(THREADS), 0, stream>>>(
        x, wswz, b_proj, bias_tab, out);
}
